// MaskedMLP_33320356282462
// MI455X (gfx1250) — matrix operand order repaired and checked by reading, not compiled, not run on hardware
//
#include <hip/hip_runtime.h>
#include <stdint.h>

typedef __attribute__((ext_vector_type(16))) __bf16 v16bf;
typedef __attribute__((ext_vector_type(8)))  float  v8f;
typedef unsigned short u16;
typedef unsigned int   u32;

#define N_EMBD 768
#define FULL   12288
#define BATCH  8
#define SEQ    1024
#define KSEL   3072

#if __has_builtin(__builtin_amdgcn_global_load_async_to_lds_b128)
#define HAVE_ASYNC_LDS 1
#endif

// ---------- helpers ----------
__device__ __forceinline__ u16 f2bf(float f) {          // native RNE convert
  union { __bf16 h; u16 u; } r;
  r.h = (__bf16)f;
  return r.u;
}
__device__ __forceinline__ u32 pack2bf(float a, float b) {
  typedef __attribute__((ext_vector_type(2))) __bf16 v2bf;
  union { v2bf v; u32 u; } r;
  r.v[0] = (__bf16)a;
  r.v[1] = (__bf16)b;
  return r.u;
}
// order-preserving float -> uint key (larger float => larger key)
__device__ __forceinline__ u32 orderKey(float f) {
  u32 u = __float_as_uint(f);
  return u ^ ((u >> 31) ? 0xFFFFFFFFu : 0x80000000u);
}

#ifdef HAVE_ASYNC_LDS
// memory -> LDS direct async copy (ASYNCcnt). Builtin expects int4-typed
// pointers (per compiler diagnostic): global (AS1) src, LDS (AS3) dst.
// Generic LDS pointers carry the LDS byte offset in their low 32 bits
// (flat aperture rule), so the truncating integer round-trip is valid.
typedef int v4i_vs __attribute__((__vector_size__(16)));
typedef __attribute__((address_space(1))) v4i_vs* as1_v4i_p;
typedef __attribute__((address_space(3))) v4i_vs* as3_v4i_p;

__device__ __forceinline__ void async_cp_b128(const void* g, void* l) {
  __builtin_amdgcn_global_load_async_to_lds_b128(
      (as1_v4i_p)(uintptr_t)g, (as3_v4i_p)(u32)(uintptr_t)l, 0, 0);
}
__device__ __forceinline__ void wait_async0() {
  asm volatile("s_wait_asynccnt 0x0" ::: "memory");
}
#endif

// ---------- control net ----------
__global__ void pool_kernel(const float* __restrict__ x, float* __restrict__ pooled) {
  int b = blockIdx.y;
  int d = blockIdx.x * 256 + threadIdx.x;                 // 3*256 = 768
  const float* xp = x + (size_t)b * SEQ * N_EMBD + d;
  float s = 0.f;
  for (int t = 0; t < SEQ; ++t) s += xp[(size_t)t * N_EMBD];
  pooled[b * N_EMBD + d] = s * (1.0f / SEQ);
}

__global__ void cvt_kernel(const float* __restrict__ x, u16* __restrict__ xb) {
  size_t i = (size_t)blockIdx.x * 256 + threadIdx.x;      // one float4 per thread
  float4 v = ((const float4*)x)[i];
  uint2 o;
  o.x = pack2bf(v.x, v.y);
  o.y = pack2bf(v.z, v.w);
  ((uint2*)xb)[i] = o;
}

__global__ void fc1_kernel(const float* __restrict__ pooled,
                           const float* __restrict__ w1, float* __restrict__ h1) {
  int b = blockIdx.x, j = threadIdx.x;
  if (j < N_EMBD / 16) {                                  // 48 outputs
    const float* p = pooled + b * N_EMBD;
    const float* w = w1 + j * N_EMBD;
    float s = 0.f;
    for (int d = 0; d < N_EMBD; ++d) s += p[d] * w[d];
    h1[b * 48 + j] = fmaxf(s, 0.f);
  }
}

__global__ void fc2_kernel(const float* __restrict__ h1,
                           const float* __restrict__ w2, float* __restrict__ ck) {
  __shared__ float hs[48];
  int b = blockIdx.y;
  if (threadIdx.x < 48) hs[threadIdx.x] = h1[b * 48 + threadIdx.x];
  __syncthreads();
  int f = blockIdx.x * 256 + threadIdx.x;                 // 48*256 = 12288
  const float* w = w2 + (size_t)f * 48;
  float s = 0.f;
  #pragma unroll
  for (int j = 0; j < 48; ++j) s += hs[j] * w[j];
  ck[(size_t)b * FULL + f] = s;
}

// ---------- exact top-K via MSB-first radix select + ordered compaction ----------
__global__ void topk_kernel(const float* __restrict__ ck, int* __restrict__ idx) {
  __shared__ u32 hist[256];
  __shared__ u32 s_prefix, s_mask, s_rem;
  __shared__ int scan[256];
  __shared__ int s_selbase, s_tiebase;

  int b = blockIdx.x, tid = threadIdx.x;
  const float* row = ck + (size_t)b * FULL;

  if (tid == 0) { s_prefix = 0u; s_mask = 0u; s_rem = KSEL; }
  __syncthreads();

  for (int shift = 24; shift >= 0; shift -= 8) {
    hist[tid] = 0u;
    __syncthreads();
    u32 pre = s_prefix, msk = s_mask;
    for (int f = tid; f < FULL; f += 256) {
      u32 k = orderKey(row[f]);
      if ((k & msk) == pre) atomicAdd(&hist[(k >> shift) & 255u], 1u);
    }
    __syncthreads();
    if (tid == 0) {
      u32 rem = s_rem;
      int d = 0;
      for (d = 255; d > 0; --d) {
        if (hist[d] >= rem) break;
        rem -= hist[d];
      }
      s_rem    = rem;
      s_prefix = pre | ((u32)d << shift);
      s_mask   = msk | (0xFFu << shift);
    }
    __syncthreads();
  }

  u32 thr = s_prefix;
  u32 rem = s_rem;                 // #elements equal to thr to include (index order)
  if (tid == 0) { s_selbase = 0; s_tiebase = 0; }
  __syncthreads();

  for (int f0 = 0; f0 < FULL; f0 += 256) {
    int f = f0 + tid;
    u32 k = orderKey(row[f]);
    int isTie = (k == thr);
    int isGt  = (k > thr);

    scan[tid] = isTie; __syncthreads();
    for (int o = 1; o < 256; o <<= 1) {
      int t = (tid >= o) ? scan[tid - o] : 0;
      __syncthreads(); scan[tid] += t; __syncthreads();
    }
    int tieExcl = scan[tid] - isTie;
    int tieTot  = scan[255];
    int tieBase = s_tiebase;
    __syncthreads();

    int inc = isGt || (isTie && (u32)(tieBase + tieExcl) < rem);

    scan[tid] = inc; __syncthreads();
    for (int o = 1; o < 256; o <<= 1) {
      int t = (tid >= o) ? scan[tid - o] : 0;
      __syncthreads(); scan[tid] += t; __syncthreads();
    }
    int selExcl = scan[tid] - inc;
    int selTot  = scan[255];
    int selBase = s_selbase;
    __syncthreads();

    if (inc) idx[(size_t)b * KSEL + selBase + selExcl] = f;
    if (tid == 0) { s_selbase = selBase + selTot; s_tiebase = tieBase + tieTot; }
    __syncthreads();
  }
}

// ---------- GEMM1: h_c[b,t,k] = relu( sum_d xb[b,t,d] * w_mfc[idx[b,k],d] ), bf16 ----------
__global__ __launch_bounds__(256) void gemm1_kernel(
    const u16* __restrict__ xb, const float* __restrict__ w_mfc,
    const int* __restrict__ idx, u16* __restrict__ hc)
{
  __shared__ __align__(16) u16 lds_a[128 * 40];   // 8-half pad: row stride 80B
  __shared__ __align__(16) u16 lds_b[128 * 40];

  const int b  = blockIdx.z;
  const int t0 = blockIdx.y * 128;
  const int k0 = blockIdx.x * 128;
  const int tid  = threadIdx.x;
  const int lane = tid & 31, w = tid >> 5;
  const int wm = w >> 2, wn = w & 3;              // wave grid 2(M) x 4(N)
  const int g  = lane >> 4, lr = lane & 15;
  const int col = tid & 15, rb = tid >> 4;
  const int c4 = tid & 3,  r4 = tid >> 2;         // A-tile uint4 loader mapping

  int gidx[8];
  #pragma unroll
  for (int i = 0; i < 8; ++i) gidx[i] = idx[b * KSEL + k0 + rb + 16 * i];

  v8f acc[4][2];
  const v8f vzero = {0.f, 0.f, 0.f, 0.f, 0.f, 0.f, 0.f, 0.f};
  #pragma unroll
  for (int mt = 0; mt < 4; ++mt)
    #pragma unroll
    for (int nt = 0; nt < 2; ++nt) acc[mt][nt] = vzero;

  const uint4* xq = (const uint4*)xb + (size_t)b * SEQ * (N_EMBD / 8);
  u32*   lb = (u32*)lds_b;
  uint4* laq = (uint4*)lds_a;

  for (int d0 = 0; d0 < N_EMBD; d0 += 32) {
    // A tile: 128 rows x 32 halfs = 128 rows x 4 uint4, 2 per thread
    #pragma unroll
    for (int i = 0; i < 2; ++i) {
      int row = r4 + 64 * i;
      const uint4* src = xq + (size_t)(t0 + row) * (N_EMBD / 8) + (d0 >> 3) + c4;
#ifdef HAVE_ASYNC_LDS
      async_cp_b128(src, &laq[row * 5 + c4]);
#else
      laq[row * 5 + c4] = *src;
#endif
    }
    // B tile: gathered w_mfc rows, f32 -> bf16 on the fly
    #pragma unroll
    for (int i = 0; i < 8; ++i) {
      int row = rb + 16 * i;
      const float* wr = w_mfc + (size_t)gidx[i] * N_EMBD + d0 + col * 2;
      lb[row * 20 + col] = pack2bf(wr[0], wr[1]);
    }
#ifdef HAVE_ASYNC_LDS
    wait_async0();
#endif
    __syncthreads();

    union { uint4 u[2]; v16bf v; } af[4], bfrag[2];
    #pragma unroll
    for (int mt = 0; mt < 4; ++mt) {
      int m = wm * 64 + mt * 16 + lr;
      af[mt].u[0] = *(const uint4*)&lds_a[m * 40 + g * 8];        // K {0..7}|{8..15}
      af[mt].u[1] = *(const uint4*)&lds_a[m * 40 + 16 + g * 8];   // K {16..23}|{24..31}
    }
    #pragma unroll
    for (int nt = 0; nt < 2; ++nt) {
      int n = wn * 32 + nt * 16 + lr;
      bfrag[nt].u[0] = *(const uint4*)&lds_b[n * 40 + g * 8];     // K {0..7}|{8..15}
      bfrag[nt].u[1] = *(const uint4*)&lds_b[n * 40 + 16 + g * 8];
    }
    #pragma unroll
    for (int mt = 0; mt < 4; ++mt)
      #pragma unroll
      for (int nt = 0; nt < 2; ++nt)
        acc[mt][nt] = __builtin_amdgcn_wmma_f32_16x16x32_bf16(
            false, af[mt].v, false, bfrag[nt].v, (short)0, acc[mt][nt], false, false);
    __syncthreads();
  }

  // relu + bf16 store (D layout: VGPR r -> M = r + 8*g, N = lr)
  #pragma unroll
  for (int mt = 0; mt < 4; ++mt)
    #pragma unroll
    for (int nt = 0; nt < 2; ++nt)
      #pragma unroll
      for (int r = 0; r < 8; ++r) {
        int t = t0 + wm * 64 + mt * 16 + r + 8 * g;
        int k = k0 + wn * 32 + nt * 16 + lr;
        hc[((size_t)b * SEQ + t) * KSEL + k] = f2bf(fmaxf(acc[mt][nt][r], 0.f));
      }
}

// ---------- GEMM2: out[b,t,e] = sum_k hc[b,t,k] * w_proj[e, idx[b,k]] ----------
__global__ __launch_bounds__(256) void gemm2_kernel(
    const u16* __restrict__ hc, const float* __restrict__ w_proj,
    const int* __restrict__ idx, float* __restrict__ out)
{
  __shared__ __align__(16) u16 lds_a[128 * 40];
  __shared__ __align__(16) u16 lds_b[128 * 40];

  const int b  = blockIdx.z;
  const int t0 = blockIdx.y * 128;
  const int e0 = blockIdx.x * 128;
  const int tid  = threadIdx.x;
  const int lane = tid & 31, w = tid >> 5;
  const int wm = w >> 2, wn = w & 3;
  const int g  = lane >> 4, lr = lane & 15;
  const int col = tid & 15, rb = tid >> 4;
  const int c4 = tid & 3,  r4 = tid >> 2;

  v8f acc[4][2];
  const v8f vzero = {0.f, 0.f, 0.f, 0.f, 0.f, 0.f, 0.f, 0.f};
  #pragma unroll
  for (int mt = 0; mt < 4; ++mt)
    #pragma unroll
    for (int nt = 0; nt < 2; ++nt) acc[mt][nt] = vzero;

  const uint4* aq = (const uint4*)hc + (size_t)b * SEQ * (KSEL / 8);
  u32*   lb = (u32*)lds_b;
  uint4* laq = (uint4*)lds_a;

  for (int k0 = 0; k0 < KSEL; k0 += 32) {
    int i0 = idx[b * KSEL + k0 + col * 2];
    int i1 = idx[b * KSEL + k0 + col * 2 + 1];
    #pragma unroll
    for (int i = 0; i < 2; ++i) {
      int row = r4 + 64 * i;
      const uint4* src = aq + (size_t)(t0 + row) * (KSEL / 8) + (k0 >> 3) + c4;
#ifdef HAVE_ASYNC_LDS
      async_cp_b128(src, &laq[row * 5 + c4]);
#else
      laq[row * 5 + c4] = *src;
#endif
    }
    // B tile: columns of w_proj gathered by idx (L2-resident), f32 -> bf16
    #pragma unroll
    for (int i = 0; i < 8; ++i) {
      int e = rb + 16 * i;
      const float* wr = w_proj + (size_t)(e0 + e) * FULL;
      lb[e * 20 + col] = pack2bf(wr[i0], wr[i1]);
    }
#ifdef HAVE_ASYNC_LDS
    wait_async0();
#endif
    __syncthreads();

    union { uint4 u[2]; v16bf v; } af[4], bfrag[2];
    #pragma unroll
    for (int mt = 0; mt < 4; ++mt) {
      int m = wm * 64 + mt * 16 + lr;
      af[mt].u[0] = *(const uint4*)&lds_a[m * 40 + g * 8];
      af[mt].u[1] = *(const uint4*)&lds_a[m * 40 + 16 + g * 8];
    }
    #pragma unroll
    for (int nt = 0; nt < 2; ++nt) {
      int n = wn * 32 + nt * 16 + lr;
      bfrag[nt].u[0] = *(const uint4*)&lds_b[n * 40 + g * 8];
      bfrag[nt].u[1] = *(const uint4*)&lds_b[n * 40 + 16 + g * 8];
    }
    #pragma unroll
    for (int mt = 0; mt < 4; ++mt)
      #pragma unroll
      for (int nt = 0; nt < 2; ++nt)
        acc[mt][nt] = __builtin_amdgcn_wmma_f32_16x16x32_bf16(
            false, af[mt].v, false, bfrag[nt].v, (short)0, acc[mt][nt], false, false);
    __syncthreads();
  }

  #pragma unroll
  for (int mt = 0; mt < 4; ++mt)
    #pragma unroll
    for (int nt = 0; nt < 2; ++nt)
      #pragma unroll
      for (int r = 0; r < 8; ++r) {
        int t = t0 + wm * 64 + mt * 16 + r + 8 * g;
        int e = e0 + wn * 32 + nt * 16 + lr;
        out[((size_t)b * SEQ + t) * N_EMBD + e] = acc[mt][nt][r];
      }
}

// ---------- launch ----------
extern "C" void kernel_launch(void* const* d_in, const int* in_sizes, int n_in,
                              void* d_out, int out_size, void* d_ws, size_t ws_size,
                              hipStream_t stream) {
  const float* x      = (const float*)d_in[0];
  const float* w1     = (const float*)d_in[1];
  const float* w2     = (const float*)d_in[2];
  const float* w_mfc  = (const float*)d_in[3];
  const float* w_proj = (const float*)d_in[4];
  float* out = (float*)d_out;

  char* ws = (char*)d_ws;
  // ws layout (all offsets 16B aligned), total ~63.5 MB
  float* pooled = (float*)(ws + 0);                 //  8*768*4      = 24576
  float* h1     = (float*)(ws + 24576);             //  8*48*4       = 1536
  float* ck     = (float*)(ws + 26112);             //  8*12288*4    = 393216
  int*   idx    = (int*)  (ws + 419328);            //  8*3072*4     = 98304
  u16*   xb     = (u16*)  (ws + 517632);            //  8*1024*768*2 = 12582912
  u16*   hc     = (u16*)  (ws + 13100544);          //  8*1024*3072*2= 50331648

  pool_kernel<<<dim3(3, BATCH), 256, 0, stream>>>(x, pooled);
  cvt_kernel <<<dim3(6144),     256, 0, stream>>>(x, xb);
  fc1_kernel <<<dim3(BATCH),     64, 0, stream>>>(pooled, w1, h1);
  fc2_kernel <<<dim3(48, BATCH),256, 0, stream>>>(h1, w2, ck);
  topk_kernel<<<dim3(BATCH),    256, 0, stream>>>(ck, idx);
  gemm1_kernel<<<dim3(KSEL / 128, SEQ / 128, BATCH), 256, 0, stream>>>(xb, w_mfc, idx, hc);
  gemm2_kernel<<<dim3(N_EMBD / 128, SEQ / 128, BATCH), 256, 0, stream>>>(hc, w_proj, idx, out);
}
